// Attention_8847632629911
// MI455X (gfx1250) — matrix operand order repaired and checked by reading, not compiled, not run on hardware
//
#include <hip/hip_runtime.h>

typedef __attribute__((ext_vector_type(16))) _Float16 v16h;
typedef __attribute__((ext_vector_type(8)))  _Float16 v8h;
typedef __attribute__((ext_vector_type(8)))  float    v8f;
typedef __attribute__((ext_vector_type(4)))  float    vf4;
typedef __attribute__((ext_vector_type(4)))  int      v4i;

#define B_ 4
#define H_ 16
#define S_ 2048
#define D_ 128
#define KV_TILE 32
#define KSTR 136   // K tile row stride in halves (128 + 8 pad)
#define VSTR 40    // V^T tile row stride in halves (32 + 8 pad)
#define PSTR 24    // P tile column stride in halves (16 + 8 pad)

// ---- CDNA5 async global->LDS path (guarded; falls back to sync staging) ----
#if defined(__has_builtin)
#if __has_builtin(__builtin_amdgcn_global_load_async_to_lds_b128)
#define USE_ASYNC 1
#define ASYNC_CP128(g, l) \
    __builtin_amdgcn_global_load_async_to_lds_b128((v4i*)(g), (v4i*)(l), 0, 0)
#endif
#endif
#ifndef USE_ASYNC
#define USE_ASYNC 0
#endif

#if defined(__has_builtin)
#if __has_builtin(__builtin_amdgcn_s_wait_asynccnt)
#define WAIT_ASYNC() __builtin_amdgcn_s_wait_asynccnt(0)
#else
#define WAIT_ASYNC() asm volatile("s_wait_asynccnt 0" ::: "memory")
#endif
#else
#define WAIT_ASYNC() asm volatile("s_wait_asynccnt 0" ::: "memory")
#endif

// raw v_exp_f32 (base-2); libm exp2f adds range-fixup VALU we don't need
#if defined(__has_builtin)
#if __has_builtin(__builtin_amdgcn_exp2f)
#define EXP2F(x) __builtin_amdgcn_exp2f(x)
#else
#define EXP2F(x) exp2f(x)
#endif
#else
#define EXP2F(x) exp2f(x)
#endif

#define WMMA_F16(A, Bv, C) \
    __builtin_amdgcn_wmma_f32_16x16x32_f16(false, (A), false, (Bv), (short)0, (C), false, false)

__global__ __launch_bounds__(128)
void fa_fwd(const float* __restrict__ Q, const float* __restrict__ Kp,
            const float* __restrict__ Vp, float* __restrict__ O) {
#if USE_ASYNC
    __shared__ __align__(16) float Ka[2][KV_TILE * D_];   // raw f32 landing slots
    __shared__ __align__(16) float Va[2][KV_TILE * D_];
#endif
    __shared__ __align__(16) _Float16 Kh[KV_TILE * KSTR];     // K block, row-major f16
    __shared__ __align__(16) _Float16 Vt[D_ * VSTR];          // V block, transposed f16
    __shared__ __align__(16) _Float16 Pl[8 * KV_TILE * PSTR]; // per (wave, q-tile) P bounce

    const int tid  = threadIdx.x;
    const int lane = tid & 31;
    const int wave = tid >> 5;
    const int n    = lane & 15;     // column in B/C layouts, row in A layout
    const int hi   = lane >> 4;     // half-wave select

    const int bh = blockIdx.x >> 4;          // 16 query-chunks of 128 per (b,h)
    const int qc = blockIdx.x & 15;
    const int q0 = qc * 128 + wave * 32;     // two 16-row tiles: q0, q0+16

    const float* qb = Q  + (size_t)bh * S_ * D_;
    const float* kb = Kp + (size_t)bh * S_ * D_;
    const float* vb = Vp + (size_t)bh * S_ * D_;
    float*       ob = O  + (size_t)bh * S_ * D_;

    // staging work split (shared by async issue and convert pass)
    const int krow = tid >> 2;          // 0..31
    const int kd   = (tid & 3) * 32;    // 0,32,64,96
    const int kpr  = (tid & 15) * 2;    // key pair for V-transpose
    const int dv   = (tid >> 4) * 16;   // dim group for V-transpose

    // ---- preload both Q tiles as A-fragments; fold softmax scale AND log2(e)
    const float scale = 0.12750277364588206f; // (1/sqrt(128)) * log2(e)
    v16h aq[2][4];
#pragma unroll
    for (int qt = 0; qt < 2; ++qt) {
        const float* qr = qb + (size_t)(q0 + qt * 16 + n) * D_;
#pragma unroll
        for (int c = 0; c < 4; ++c) {
            const float* p0 = qr + 32 * c + 8 * hi;
            vf4 x0 = *(const vf4*)(p0);
            vf4 x1 = *(const vf4*)(p0 + 4);
            vf4 y0 = *(const vf4*)(p0 + 16);
            vf4 y1 = *(const vf4*)(p0 + 20);
#pragma unroll
            for (int i = 0; i < 4; ++i) {
                aq[qt][c][i]      = (_Float16)(x0[i] * scale);
                aq[qt][c][4 + i]  = (_Float16)(x1[i] * scale);
                aq[qt][c][8 + i]  = (_Float16)(y0[i] * scale);
                aq[qt][c][12 + i] = (_Float16)(y1[i] * scale);
            }
        }
    }

    // ones B-fragment: row-sum of P via the matrix pipe (ℓ accumulator)
    v16h vones;
#pragma unroll
    for (int i = 0; i < 16; ++i) vones[i] = (_Float16)1.0f;

    // ---- running state: O accumulators, ℓ accumulators (WMMA), row max
    v8f o[2][8];
    v8f osum[2];
    float mrun[2][8];
#pragma unroll
    for (int qt = 0; qt < 2; ++qt) {
#pragma unroll
        for (int t = 0; t < 8; ++t)
#pragma unroll
            for (int r = 0; r < 8; ++r) o[qt][t][r] = 0.0f;
#pragma unroll
        for (int r = 0; r < 8; ++r) { osum[qt][r] = 0.0f; mrun[qt][r] = -3.0e38f; }
    }

#if USE_ASYNC
    // prologue: async-stage block 0 into slot 0
    {
        const float* gk = kb + (size_t)krow * D_ + kd;
        const float* gv = vb + (size_t)krow * D_ + kd;
        float* lk = &Ka[0][krow * D_ + kd];
        float* lv = &Va[0][krow * D_ + kd];
#pragma unroll
        for (int j = 0; j < 8; ++j) {
            ASYNC_CP128(gk + 4 * j, lk + 4 * j);
            ASYNC_CP128(gv + 4 * j, lv + 4 * j);
        }
    }
#endif

    for (int it = 0; it < S_ / KV_TILE; ++it) {
        const int kv0 = it * KV_TILE;
#if USE_ASYNC
        const int cur = it & 1;
        WAIT_ASYNC();
#endif
        __syncthreads();   // (A) prev compute done; async slot `cur` visible everywhere

        // ---- convert pass: K -> row-major f16 LDS, V -> transposed f16 LDS
        {
#if USE_ASYNC
            const float* ksrc = &Ka[cur][krow * D_ + kd];
            const float* s0   = &Va[cur][kpr * D_ + dv];
            const float* s1   = s0 + D_;
#else
            const float* ksrc = kb + (size_t)(kv0 + krow) * D_ + kd;
            const float* s0   = vb + (size_t)(kv0 + kpr) * D_ + dv;
            const float* s1   = s0 + D_;
#endif
#pragma unroll
            for (int g = 0; g < 4; ++g) {
                vf4 a = *(const vf4*)(ksrc + 8 * g);
                vf4 b = *(const vf4*)(ksrc + 8 * g + 4);
                v8h h8;
#pragma unroll
                for (int i = 0; i < 4; ++i) { h8[i] = (_Float16)a[i]; h8[4 + i] = (_Float16)b[i]; }
                *(v8h*)&Kh[krow * KSTR + kd + 8 * g] = h8;
            }
            float a0[16], a1[16];
#pragma unroll
            for (int g = 0; g < 4; ++g) {
                vf4 x = *(const vf4*)(s0 + 4 * g);
                vf4 y = *(const vf4*)(s1 + 4 * g);
#pragma unroll
                for (int i = 0; i < 4; ++i) { a0[4 * g + i] = x[i]; a1[4 * g + i] = y[i]; }
            }
#pragma unroll
            for (int j = 0; j < 16; ++j) {
                union { _Float16 h[2]; unsigned u; } pk;
                pk.h[0] = (_Float16)a0[j];
                pk.h[1] = (_Float16)a1[j];
                *(unsigned*)&Vt[(dv + j) * VSTR + kpr] = pk.u;
            }
        }
        __syncthreads();   // (B) f16 tiles ready; f32 slot fully consumed

#if USE_ASYNC
        // ---- async-stage the NEXT block into the other slot while we compute
        if (kv0 + KV_TILE < S_) {
            const int nxt = cur ^ 1;
            const float* gk = kb + (size_t)(kv0 + KV_TILE + krow) * D_ + kd;
            const float* gv = vb + (size_t)(kv0 + KV_TILE + krow) * D_ + kd;
            float* lk = &Ka[nxt][krow * D_ + kd];
            float* lv = &Va[nxt][krow * D_ + kd];
#pragma unroll
            for (int j = 0; j < 8; ++j) {
                ASYNC_CP128(gk + 4 * j, lk + 4 * j);
                ASYNC_CP128(gv + 4 * j, lv + 4 * j);
            }
        }
#else
        if (kv0 + KV_TILE < S_) {
            __builtin_prefetch(kb + (size_t)(kv0 + KV_TILE + krow) * D_ + kd, 0, 0);
            __builtin_prefetch(vb + (size_t)(kv0 + KV_TILE + krow) * D_ + kd, 0, 0);
        }
#endif

        // ---- hoist ALL K fragments first: 16 ds_load_b128 in flight, then
        //      drain into the WMMA chain with partial dscnt waits
        v16h bkf[2][4];
#pragma unroll
        for (int nt = 0; nt < 2; ++nt) {
            const int kr = nt * 16 + n;
#pragma unroll
            for (int c = 0; c < 4; ++c) {
                const v8h* bp = (const v8h*)&Kh[kr * KSTR + 32 * c + 8 * hi];
                v8h blo = bp[0], bhi = bp[2];
                bkf[nt][c] = __builtin_shufflevector(blo, bhi, 0, 1, 2, 3, 4, 5, 6, 7,
                                                     8, 9, 10, 11, 12, 13, 14, 15);
            }
        }

        // ---- S = Q K^T : 2 q-tiles x 2 key-tiles; each bk fragment feeds 2 WMMAs
        v8f sc[2][2];
#pragma unroll
        for (int nt = 0; nt < 2; ++nt) {
#pragma unroll
            for (int qt = 0; qt < 2; ++qt)
#pragma unroll
                for (int r = 0; r < 8; ++r) sc[qt][nt][r] = 0.0f;
#pragma unroll
            for (int c = 0; c < 4; ++c) {
                sc[0][nt] = WMMA_F16(aq[0][c], bkf[nt][c], sc[0][nt]);
                sc[1][nt] = WMMA_F16(aq[1][c], bkf[nt][c], sc[1][nt]);
            }
        }

        // ---- online softmax (base-2) + P bounce per q-tile
        v16h ap[2];
#pragma unroll
        for (int qt = 0; qt < 2; ++qt) {
            v8h ph0, ph1;
#pragma unroll
            for (int r = 0; r < 8; ++r) {
                float s0 = sc[qt][0][r];
                float s1 = sc[qt][1][r];
                float mx = fmaxf(s0, s1);
#pragma unroll
                for (int msk = 1; msk < 16; msk <<= 1) mx = fmaxf(mx, __shfl_xor(mx, msk, 16));
                float mnew = fmaxf(mrun[qt][r], mx);
                float al   = EXP2F(mrun[qt][r] - mnew);
                float p0   = EXP2F(s0 - mnew);
                float p1   = EXP2F(s1 - mnew);
                mrun[qt][r] = mnew;
                ph0[r] = (_Float16)p0;
                ph1[r] = (_Float16)p1;
                osum[qt][r] *= al;                        // rescale ℓ like O
#pragma unroll
                for (int t = 0; t < 8; ++t) o[qt][t][r] *= al;
            }
            _Float16* pw = &Pl[(wave * 2 + qt) * (KV_TILE * PSTR)];
            *(v8h*)&pw[n * PSTR + 8 * hi]        = ph0;   // column-major store
            *(v8h*)&pw[(16 + n) * PSTR + 8 * hi] = ph1;
#pragma unroll
            for (int h = 0; h < 16; ++h) {                // gather back in A-layout
                const int k = h + (h & 8) + 8 * hi;
                ap[qt][h] = pw[k * PSTR + n];
            }
        }

        // ---- ℓ += P * 1s (row sums on the matrix pipe, replaces sum butterfly)
        osum[0] = WMMA_F16(ap[0], vones, osum[0]);
        osum[1] = WMMA_F16(ap[1], vones, osum[1]);

        // ---- O += P V : each bv fragment feeds 2 WMMAs
#pragma unroll
        for (int t = 0; t < 8; ++t) {
            const v8h* vp = (const v8h*)&Vt[(16 * t + n) * VSTR + 8 * hi];
            v8h blo = vp[0], bhi = vp[2];
            v16h bv = __builtin_shufflevector(blo, bhi, 0, 1, 2, 3, 4, 5, 6, 7,
                                              8, 9, 10, 11, 12, 13, 14, 15);
            o[0][t] = WMMA_F16(ap[0], bv, o[0][t]);
            o[1][t] = WMMA_F16(ap[1], bv, o[1][t]);
        }
    }

    // ---- normalize and write back (every lane holds its row's ℓ in osum)
#pragma unroll
    for (int qt = 0; qt < 2; ++qt) {
#pragma unroll
        for (int r = 0; r < 8; ++r) {
            const float inv = 1.0f / osum[qt][r];
            float* orow = ob + (size_t)(q0 + qt * 16 + r + 8 * hi) * D_;
#pragma unroll
            for (int t = 0; t < 8; ++t) orow[16 * t + n] = o[qt][t][r] * inv;
        }
    }
}

extern "C" void kernel_launch(void* const* d_in, const int* in_sizes, int n_in,
                              void* d_out, int out_size, void* d_ws, size_t ws_size,
                              hipStream_t stream) {
    const float* q = (const float*)d_in[0];
    const float* k = (const float*)d_in[1];
    const float* v = (const float*)d_in[2];
    float* out = (float*)d_out;
    dim3 grid(B_ * H_ * (S_ / 128));
    dim3 block(128);
    fa_fwd<<<grid, block, 0, stream>>>(q, k, v, out);
    (void)in_sizes; (void)n_in; (void)out_size; (void)d_ws; (void)ws_size;
}
